// SchNetEncoder_64888365908356
// MI455X (gfx1250) — compile-verified, matrix operand order checked by reading, not run on hardware
//
#include <hip/hip_runtime.h>
#include <hip/hip_bf16.h>
#include <math.h>

typedef __attribute__((ext_vector_type(16))) _Float16 v16h;
typedef __attribute__((ext_vector_type(8)))  float    v8f;

// ---------------------------------------------------------------------------
// Constants for this problem instance
// ---------------------------------------------------------------------------
#define BGRAPH 32
#define NNODE  512
#define MNBR   32
#define DEDGE  64
#define HDIM   128
#define DOUT   64
#define NLAYER 3

// Fast tanh: gfx1250 has a native V_TANH_F32 TRANS instruction.
__device__ inline float fast_tanh(float x) {
#if defined(__gfx1250__) && __has_builtin(__builtin_amdgcn_tanhf)
  return __builtin_amdgcn_tanhf(x);
#else
  return tanhf(x);
#endif
}

// ---------------------------------------------------------------------------
// Weight pre-pack: f32 [K, HD] row-major -> f16 B-fragment order.
// For each (kt, nt) 32x16 tile, 32 lanes x 16 contiguous halfs:
//   dst[((kt*NT + nt)*32 + lane)*16 + j] = W[kt*32 + kofs(j,lane)][nt*16 + lane%16]
//   kofs(j,lane) = ((j>>3)<<4) + ((lane>>4)<<3) + (j&7)
// ---------------------------------------------------------------------------
__global__ void pack_b_kernel(const float* __restrict__ W, _Float16* __restrict__ dst,
                              int K, int HD) {
  int idx = blockIdx.x * blockDim.x + threadIdx.x;  // one per (tile, lane)
  int NT = HD >> 4;
  int total = (K >> 5) * NT * 32;
  if (idx >= total) return;
  int lane = idx & 31;
  int tile = idx >> 5;
  int nt = tile % NT;
  int kt = tile / NT;
  int col  = nt * 16 + (lane & 15);
  int half = lane >> 4;
  _Float16* o = dst + (size_t)idx * 16;
#pragma unroll
  for (int j = 0; j < 16; ++j) {
    int k = kt * 32 + ((j >> 3) << 4) + (half << 3) + (j & 7);
    o[j] = (_Float16)W[(size_t)k * HD + col];
  }
}

// Load an A fragment from LDS.  ap points at &As[row*LDA + kt*32 + half*8]
// (16-byte aligned by construction: LDA is a multiple of 8 halfs).
__device__ inline v16h load_afrag(const _Float16* ap) {
  v16h a;
#pragma unroll
  for (int j = 0; j < 8; ++j) { a[j] = ap[j]; a[j + 8] = ap[16 + j]; }
  return a;
}

// ---------------------------------------------------------------------------
// Generic tiled GEMM:  out[R, HD] = act(A[R, K] @ Wpacked + bias) (+ res)
// Block = 128 threads (4 waves), each block computes 32 rows x HD cols.
// ---------------------------------------------------------------------------
template <int K, int HD>
__global__ __launch_bounds__(128) void gemm_kernel(
    const float* __restrict__ A, const _Float16* __restrict__ Bp,
    const float* __restrict__ bias, const float* __restrict__ res,
    float* __restrict__ out, int act) {
  constexpr int NT  = HD / 16;   // N tiles (8 or 4)
  constexpr int KT  = K / 32;    // K steps (2 or 4)
  constexpr int TPW = NT / 4;    // N tiles per wave (2 or 1)
  constexpr int LDA = K + 8;     // halfs; row stride stays 16B-aligned

  __shared__ __align__(16) _Float16 As[32 * LDA];

  const int tid  = threadIdx.x;
  const int wave = tid >> 5;
  const int lane = tid & 31;
  const int half = lane >> 4;
  const int l16  = lane & 15;
  const size_t rowbase = (size_t)blockIdx.x * 32;

  // Prefetch packed weights (L2-resident; pulls line ahead of first K-step).
  __builtin_prefetch(Bp + (size_t)tid * (K * HD / 128), 0, 3);

  // Stage A tile, converting f32 -> f16 (coalesced).
  for (int i = tid; i < 32 * K; i += 128) {
    int r = i / K, c = i % K;
    As[r * LDA + c] = (_Float16)A[(rowbase + r) * K + c];
  }
  __syncthreads();

  v8f zero8 = {0.f, 0.f, 0.f, 0.f, 0.f, 0.f, 0.f, 0.f};
  v8f acc[2][TPW];
#pragma unroll
  for (int mt = 0; mt < 2; ++mt)
#pragma unroll
    for (int t = 0; t < TPW; ++t) acc[mt][t] = zero8;

#pragma unroll
  for (int kt = 0; kt < KT; ++kt) {
    v16h af[2];
#pragma unroll
    for (int mt = 0; mt < 2; ++mt)
      af[mt] = load_afrag(&As[(mt * 16 + l16) * LDA + kt * 32 + half * 8]);
#pragma unroll
    for (int t = 0; t < TPW; ++t) {
      int nt = wave * TPW + t;
      v16h bf = *(const v16h*)(Bp + (((size_t)(kt * NT + nt)) * 32 + lane) * 16);
#pragma unroll
      for (int mt = 0; mt < 2; ++mt)
        acc[mt][t] = __builtin_amdgcn_wmma_f32_16x16x32_f16(
            false, af[mt], false, bf, (short)0, acc[mt][t], false, false);
    }
  }

#pragma unroll
  for (int t = 0; t < TPW; ++t) {
    int nt  = wave * TPW + t;
    int col = nt * 16 + l16;
    float bv = bias ? bias[col] : 0.f;
#pragma unroll
    for (int mt = 0; mt < 2; ++mt) {
#pragma unroll
      for (int r = 0; r < 8; ++r) {
        size_t row = rowbase + mt * 16 + half * 8 + r;
        float v = acc[mt][t][r] + bv;
        if (act) v = fast_tanh(v);
        if (res) v += res[row * HD + col];
        out[row * HD + col] = v;
      }
    }
  }
}

// ---------------------------------------------------------------------------
// Fused interaction kernel — one workgroup (128 threads) per (b, n) node.
//   filt = tanh(rbf @ Wf1 + bf1) @ Wf2 + bf2          (two WMMA GEMMs in LDS)
//   nf   = gather(f0, nbr)
//   conv = nf * filt; scores = conv . attn; softmax over M; agg = sum w*conv
// Nothing of size [B,N,M,H] ever touches global memory.
// ---------------------------------------------------------------------------
__global__ __launch_bounds__(128) void interaction_kernel(
    const float* __restrict__ rbf,      // [B,N,M,DEDGE]
    const int*   __restrict__ nbr,      // [B,N,M]
    const float* __restrict__ f0,       // [B,N,H]
    const _Float16* __restrict__ Wf1p,  // packed, K=64
    const float* __restrict__ bf1,      // [H]
    const _Float16* __restrict__ Wf2p,  // packed, K=128
    const float* __restrict__ bf2,      // [H]
    const float* __restrict__ attn,     // [H]
    float* __restrict__ agg) {          // [B,N,H]
  constexpr int LDA1 = DEDGE + 8;   // 72 halfs
  constexpr int LDA2 = HDIM + 8;    // 136 halfs

  __shared__ __align__(16) _Float16 Rs[MNBR * LDA1];   // rbf tile (f16)
  __shared__ __align__(16) _Float16 Hs[MNBR * LDA2];   // tanh hidden (f16)
  __shared__ float nf[MNBR][HDIM + 4];                 // gathered f0 rows
  __shared__ float filt[MNBR][HDIM + 4];               // filter output
  __shared__ float sred[MNBR][4];
  __shared__ float sw[MNBR];
  __shared__ float sinv;

  const int node = blockIdx.x;           // b*N + n
  const int tid  = threadIdx.x;
  const int wave = tid >> 5;
  const int lane = tid & 31;
  const int half = lane >> 4;
  const int l16  = lane & 15;
  const int bidx = node / NNODE;

  // Prefetch both packed filter weight matrices (hot in L2 across 16K WGs).
  __builtin_prefetch(Wf1p + (size_t)tid * (DEDGE * HDIM / 128), 0, 3);
  __builtin_prefetch(Wf2p + (size_t)tid * (HDIM * HDIM / 128), 0, 3);

  // Stage rbf tile (32 x 64, f32 -> f16).
  const float* rb = rbf + (size_t)node * MNBR * DEDGE;
  for (int i = tid; i < MNBR * DEDGE; i += 128) {
    int r = i >> 6, c = i & 63;
    Rs[r * LDA1 + c] = (_Float16)rb[i];
  }
  // Stage gathered neighbor features (32 rows x 128, f32).
  const int* nb = nbr + (size_t)node * MNBR;
  for (int i = tid; i < MNBR * HDIM; i += 128) {
    int r = i >> 7, c = i & 127;
    int j = nb[r];
    nf[r][c] = f0[((size_t)bidx * NNODE + j) * HDIM + c];
  }
  __syncthreads();

  v8f zero8 = {0.f, 0.f, 0.f, 0.f, 0.f, 0.f, 0.f, 0.f};

  // ---- GEMM1: hidden = tanh(Rs @ Wf1 + bf1) --------------------------------
  {
    v8f acc[2][2];
#pragma unroll
    for (int mt = 0; mt < 2; ++mt) { acc[mt][0] = zero8; acc[mt][1] = zero8; }
#pragma unroll
    for (int kt = 0; kt < DEDGE / 32; ++kt) {   // 2 steps
      v16h af[2];
#pragma unroll
      for (int mt = 0; mt < 2; ++mt)
        af[mt] = load_afrag(&Rs[(mt * 16 + l16) * LDA1 + kt * 32 + half * 8]);
#pragma unroll
      for (int t = 0; t < 2; ++t) {
        int nt = wave * 2 + t;
        v16h bfr = *(const v16h*)(Wf1p + (((size_t)(kt * 8 + nt)) * 32 + lane) * 16);
#pragma unroll
        for (int mt = 0; mt < 2; ++mt)
          acc[mt][t] = __builtin_amdgcn_wmma_f32_16x16x32_f16(
              false, af[mt], false, bfr, (short)0, acc[mt][t], false, false);
      }
    }
#pragma unroll
    for (int t = 0; t < 2; ++t) {
      int col = (wave * 2 + t) * 16 + l16;
      float bv = bf1[col];
#pragma unroll
      for (int mt = 0; mt < 2; ++mt)
#pragma unroll
        for (int r = 0; r < 8; ++r)
          Hs[(mt * 16 + half * 8 + r) * LDA2 + col] =
              (_Float16)fast_tanh(acc[mt][t][r] + bv);
    }
  }
  __syncthreads();

  // ---- GEMM2: filt = Hs @ Wf2 + bf2 ---------------------------------------
  {
    v8f acc[2][2];
#pragma unroll
    for (int mt = 0; mt < 2; ++mt) { acc[mt][0] = zero8; acc[mt][1] = zero8; }
#pragma unroll
    for (int kt = 0; kt < HDIM / 32; ++kt) {    // 4 steps
      v16h af[2];
#pragma unroll
      for (int mt = 0; mt < 2; ++mt)
        af[mt] = load_afrag(&Hs[(mt * 16 + l16) * LDA2 + kt * 32 + half * 8]);
#pragma unroll
      for (int t = 0; t < 2; ++t) {
        int nt = wave * 2 + t;
        v16h bfr = *(const v16h*)(Wf2p + (((size_t)(kt * 8 + nt)) * 32 + lane) * 16);
#pragma unroll
        for (int mt = 0; mt < 2; ++mt)
          acc[mt][t] = __builtin_amdgcn_wmma_f32_16x16x32_f16(
              false, af[mt], false, bfr, (short)0, acc[mt][t], false, false);
      }
    }
#pragma unroll
    for (int t = 0; t < 2; ++t) {
      int col = (wave * 2 + t) * 16 + l16;
      float bv = bf2[col];
#pragma unroll
      for (int mt = 0; mt < 2; ++mt)
#pragma unroll
        for (int r = 0; r < 8; ++r)
          filt[mt * 16 + half * 8 + r][col] = acc[mt][t][r] + bv;
    }
  }
  __syncthreads();

  // ---- attention scores: s[m] = sum_c nf*filt*attn -------------------------
  {
    int m = tid >> 2, ch = tid & 3;
    float s = 0.f;
    int c0 = ch * 32;
#pragma unroll 8
    for (int c = c0; c < c0 + 32; ++c) s += nf[m][c] * filt[m][c] * attn[c];
    sred[m][ch] = s;
  }
  __syncthreads();
  if (tid < MNBR) sw[tid] = sred[tid][0] + sred[tid][1] + sred[tid][2] + sred[tid][3];
  __syncthreads();
  if (tid == 0) {
    float mx = sw[0];
#pragma unroll
    for (int m = 1; m < MNBR; ++m) mx = fmaxf(mx, sw[m]);
    float sum = 0.f;
#pragma unroll
    for (int m = 0; m < MNBR; ++m) { float e = __expf(sw[m] - mx); sw[m] = e; sum += e; }
    sinv = 1.f / sum;
  }
  __syncthreads();

  // ---- aggregate: agg[c] = sum_m w[m] * nf[m][c] * filt[m][c] --------------
  {
    float a = 0.f;
#pragma unroll 8
    for (int m = 0; m < MNBR; ++m) a += sw[m] * nf[m][tid] * filt[m][tid];
    agg[(size_t)node * HDIM + tid] = a * sinv;
  }
}

// ---------------------------------------------------------------------------
// Global mean pool over nodes: pooled[b][c] = mean_n h[b][n][c]
// ---------------------------------------------------------------------------
__global__ __launch_bounds__(128) void pool_kernel(const float* __restrict__ h,
                                                   float* __restrict__ pooled) {
  int b = blockIdx.x, c = threadIdx.x;
  float s = 0.f;
  for (int n = 0; n < NNODE; ++n) s += h[((size_t)b * NNODE + n) * HDIM + c];
  pooled[(size_t)b * HDIM + c] = s * (1.f / (float)NNODE);
}

// ---------------------------------------------------------------------------
// Host driver
// ---------------------------------------------------------------------------
extern "C" void kernel_launch(void* const* d_in, const int* in_sizes, int n_in,
                              void* d_out, int out_size, void* d_ws, size_t ws_size,
                              hipStream_t stream) {
  const float* x      = (const float*)d_in[0];
  const float* rbf    = (const float*)d_in[1];
  const int*   nbr    = (const int*)d_in[2];
  const float* W_emb  = (const float*)d_in[3];
  const float* b_emb  = (const float*)d_in[4];
  const float* Wi     = (const float*)d_in[5];
  const float* Wf1    = (const float*)d_in[6];
  const float* bf1    = (const float*)d_in[7];
  const float* Wf2    = (const float*)d_in[8];
  const float* bf2    = (const float*)d_in[9];
  const float* attn   = (const float*)d_in[10];
  const float* Wo1    = (const float*)d_in[11];
  const float* bo1    = (const float*)d_in[12];
  const float* Wo2    = (const float*)d_in[13];
  const float* bo2    = (const float*)d_in[14];
  const float* Wp1    = (const float*)d_in[15];
  const float* bp1    = (const float*)d_in[16];
  const float* Wp2    = (const float*)d_in[17];
  const float* bp2    = (const float*)d_in[18];
  float* out = (float*)d_out;

  // ---- workspace layout ----------------------------------------------------
  char* w = (char*)d_ws;
  size_t off = 0;
  auto nexth = [&](size_t nelem) -> _Float16* {
    _Float16* p = (_Float16*)(w + off);
    off = (off + nelem * sizeof(_Float16) + 255) & ~(size_t)255;
    return p;
  };
  auto nextf = [&](size_t nelem) -> float* {
    float* p = (float*)(w + off);
    off = (off + nelem * sizeof(float) + 255) & ~(size_t)255;
    return p;
  };

  _Float16* W_emb_p = nexth((size_t)DEDGE * HDIM);
  _Float16* Wi_p    = nexth((size_t)NLAYER * HDIM * HDIM);
  _Float16* Wf1_p   = nexth((size_t)NLAYER * DEDGE * HDIM);
  _Float16* Wf2_p   = nexth((size_t)NLAYER * HDIM * HDIM);
  _Float16* Wo1_p   = nexth((size_t)NLAYER * HDIM * HDIM);
  _Float16* Wo2_p   = nexth((size_t)NLAYER * HDIM * HDIM);
  _Float16* Wp1_p   = nexth((size_t)HDIM * HDIM);
  _Float16* Wp2_p   = nexth((size_t)HDIM * DOUT);

  const size_t RN = (size_t)BGRAPH * NNODE;  // 16384 rows
  float* hbuf   = nextf(RN * HDIM);
  float* f0buf  = nextf(RN * HDIM);          // also reused for t1
  float* aggbuf = nextf(RN * HDIM);
  float* pooled = nextf((size_t)BGRAPH * HDIM);
  float* p1buf  = nextf((size_t)BGRAPH * HDIM);
  (void)ws_size; (void)n_in; (void)in_sizes; (void)out_size;

  // ---- 1. pack all weights into f16 B-fragment order ----------------------
  auto pack = [&](const float* Wsrc, _Float16* dst, int K, int HD) {
    int total = (K / 32) * (HD / 16) * 32;
    pack_b_kernel<<<(total + 127) / 128, 128, 0, stream>>>(Wsrc, dst, K, HD);
  };
  pack(W_emb, W_emb_p, DEDGE, HDIM);
  pack(Wp1, Wp1_p, HDIM, HDIM);
  pack(Wp2, Wp2_p, HDIM, DOUT);
  for (int l = 0; l < NLAYER; ++l) {
    pack(Wi  + (size_t)l * HDIM * HDIM,  Wi_p  + (size_t)l * HDIM * HDIM,  HDIM,  HDIM);
    pack(Wf1 + (size_t)l * DEDGE * HDIM, Wf1_p + (size_t)l * DEDGE * HDIM, DEDGE, HDIM);
    pack(Wf2 + (size_t)l * HDIM * HDIM,  Wf2_p + (size_t)l * HDIM * HDIM,  HDIM,  HDIM);
    pack(Wo1 + (size_t)l * HDIM * HDIM,  Wo1_p + (size_t)l * HDIM * HDIM,  HDIM,  HDIM);
    pack(Wo2 + (size_t)l * HDIM * HDIM,  Wo2_p + (size_t)l * HDIM * HDIM,  HDIM,  HDIM);
  }

  const int gridBig = (int)(RN / 32);  // 512 blocks of 32 rows

  // ---- 2. embedding: h = x @ W_emb + b_emb --------------------------------
  gemm_kernel<DEDGE, HDIM><<<gridBig, 128, 0, stream>>>(
      x, W_emb_p, b_emb, nullptr, hbuf, /*act=*/0);

  // ---- 3. interaction layers ----------------------------------------------
  for (int l = 0; l < NLAYER; ++l) {
    // f0 = h @ Wi[l]
    gemm_kernel<HDIM, HDIM><<<gridBig, 128, 0, stream>>>(
        hbuf, Wi_p + (size_t)l * HDIM * HDIM, nullptr, nullptr, f0buf, 0);

    // fused cfconv + attention aggregation -> agg
    interaction_kernel<<<(int)RN, 128, 0, stream>>>(
        rbf, nbr, f0buf,
        Wf1_p + (size_t)l * DEDGE * HDIM, bf1 + (size_t)l * HDIM,
        Wf2_p + (size_t)l * HDIM * HDIM,  bf2 + (size_t)l * HDIM,
        attn + (size_t)l * HDIM, aggbuf);

    // t1 = tanh(agg @ Wo1 + bo1)   (reuse f0buf)
    gemm_kernel<HDIM, HDIM><<<gridBig, 128, 0, stream>>>(
        aggbuf, Wo1_p + (size_t)l * HDIM * HDIM, bo1 + (size_t)l * HDIM,
        nullptr, f0buf, /*act=*/1);

    // h = h + t1 @ Wo2 + bo2
    gemm_kernel<HDIM, HDIM><<<gridBig, 128, 0, stream>>>(
        f0buf, Wo2_p + (size_t)l * HDIM * HDIM, bo2 + (size_t)l * HDIM,
        /*res=*/hbuf, hbuf, /*act=*/0);
  }

  // ---- 4. mean pool + output MLP ------------------------------------------
  pool_kernel<<<BGRAPH, 128, 0, stream>>>(hbuf, pooled);
  gemm_kernel<HDIM, HDIM><<<1, 128, 0, stream>>>(
      pooled, Wp1_p, bp1, nullptr, p1buf, /*act=*/1);
  gemm_kernel<HDIM, DOUT><<<1, 128, 0, stream>>>(
      p1buf, Wp2_p, bp2, nullptr, out, /*act=*/0);
}
